// AhpcNetwork_67662914781488
// MI455X (gfx1250) — matrix operand order repaired and checked by reading, not compiled, not run on hardware
//
#include <hip/hip_runtime.h>
#include <hip/hip_bf16.h>

typedef __attribute__((ext_vector_type(16))) __bf16 v16bf;
typedef __attribute__((ext_vector_type(8)))  __bf16 v8bf;
typedef __attribute__((ext_vector_type(8)))  float  v8f;
typedef unsigned int u32x4 __attribute__((ext_vector_type(4)));
typedef int          i32x4 __attribute__((ext_vector_type(4)));
typedef int          i32x8 __attribute__((ext_vector_type(8)));

#define T_STEPS 100
#define BATCH   256
#define NIN     700
#define K1P     704     // layer-1 K padded to multiple of 32
#define H1      2048
#define H2      2048
#define NOUT    20
#define K2      4096    // H1 (s1) + H2 (spk_r) combined K
#define KC      512     // TDM A-chunk for layer-2 (8 chunks)

#define WMMA_BF16(a, b, c) \
    __builtin_amdgcn_wmma_f32_16x16x32_bf16(false, (a), false, (b), (short)0, (c), false, false)

// Two 16-byte loads -> one v16bf (works for global and LDS after inlining).
__device__ __forceinline__ v16bf load2x8(const __bf16* p0, const __bf16* p1) {
    union { v16bf v; v8bf h[2]; } u;
    u.h[0] = *(const v8bf*)p0;
    u.h[1] = *(const v8bf*)p1;
    return u.v;
}

// ---- Tensor Data Mover: 2D tile (rows x width bf16, row stride in elems) -> LDS ----
__device__ __forceinline__ void tdm_load_2d(void* lds_dst, const void* gsrc,
                                            unsigned width, unsigned rows,
                                            unsigned stride) {
    // Flat LDS pointer: low 32 bits are the LDS byte offset (aperture in high bits).
    unsigned lds_off = (unsigned)(unsigned long long)lds_dst;
    unsigned long long ga = (unsigned long long)gsrc;
    // D# group 0: count=1 | lds_addr | global_addr[56:0] | type=2 ("image")
    u32x4 g0 = { 1u,
                 lds_off,
                 (unsigned)ga,
                 (unsigned)((ga >> 32) & 0x1FFFFFFu) | (2u << 30) };
    // D# group 1: data_size=1 (2B); tensor_dim0=width, tensor_dim1=rows;
    // tile_dim0=width, tile_dim1=rows; tensor_dim0_stride=stride.
    i32x8 g1 = { (int)(1u << 16),
                 (int)((width & 0xFFFFu) << 16),
                 (int)((width >> 16) | ((rows & 0xFFFFu) << 16)),
                 (int)((rows >> 16) | ((width & 0xFFFFu) << 16)),
                 (int)(rows & 0xFFFFu),
                 (int)stride,
                 0, 0 };
    i32x4 z4 = { 0, 0, 0, 0 };
#if defined(__clang_major__) && (__clang_major__ >= 23)
    i32x8 z8 = { 0, 0, 0, 0, 0, 0, 0, 0 };
    __builtin_amdgcn_tensor_load_to_lds(g0, g1, z4, z4, z8, 0);
#else
    __builtin_amdgcn_tensor_load_to_lds(g0, g1, z4, z4, 0);
#endif
}

// ---------------- setup kernels ----------------

__global__ __launch_bounds__(256) void zero_f32(float* __restrict__ p, long n) {
    long i = (long)blockIdx.x * blockDim.x + threadIdx.x;
    long stride = (long)gridDim.x * blockDim.x;
    for (; i < n; i += stride) p[i] = 0.0f;
}

__global__ __launch_bounds__(256) void cvt_w1(const float* __restrict__ W1,
                                              __bf16* __restrict__ W1b) {
    int idx = blockIdx.x * blockDim.x + threadIdx.x;
    if (idx >= H1 * K1P) return;
    int h = idx / K1P, k = idx % K1P;
    W1b[idx] = (k < NIN) ? (__bf16)W1[h * NIN + k] : (__bf16)0.0f;
}

__global__ __launch_bounds__(256) void cvt_w2v(const float* __restrict__ W2,
                                               const float* __restrict__ V,
                                               __bf16* __restrict__ W2Vb) {
    int idx = blockIdx.x * blockDim.x + threadIdx.x;
    if (idx >= H2 * K2) return;
    int n = idx / K2, k = idx % K2;
    W2Vb[idx] = (k < H1) ? (__bf16)W2[n * H1 + k] : (__bf16)V[n * H2 + (k - H1)];
}

__global__ __launch_bounds__(256) void cvt_x(const float* __restrict__ x,
                                             __bf16* __restrict__ xT) {
    long i = (long)blockIdx.x * blockDim.x + threadIdx.x;
    if (i >= (long)T_STEPS * BATCH * K1P) return;
    int t   = (int)(i / (BATCH * K1P));
    int rem = (int)(i % (BATCH * K1P));
    int b = rem / K1P, k = rem % K1P;
    xT[i] = (k < NIN) ? (__bf16)x[(long)b * NIN * T_STEPS + (long)k * T_STEPS + t]
                      : (__bf16)0.0f;
}

// ---------------- per-timestep kernels ----------------

// Layer 1: cur1 = x_t @ W1b^T ; fused Leaky (reset-by-subtract).
// A tile (16 x 704) staged once via TDM into LDS, shared by 4 waves.
// Wave tile 16M x 64N, grid = (16, 8), 128 threads.
__global__ __launch_bounds__(128) void gemm1_neuron1(
    const __bf16* __restrict__ A,     // [BATCH][K1P] (this t)
    const __bf16* __restrict__ Bw,    // [H1][K1P]
    const float*  __restrict__ bias,
    float*        __restrict__ mem1,  // [BATCH][H1]
    __bf16*       __restrict__ s1,    // [BATCH][H1]
    const float*  __restrict__ beta_p,
    const float*  __restrict__ thr_p)
{
    __shared__ __align__(16) __bf16 As[16 * K1P];
    const int lane = threadIdx.x & 31;
    const int wave = threadIdx.x >> 5;
    const int mb = blockIdx.x * 16;
    const int nb = blockIdx.y * 256 + wave * 64;
    const int nc = lane & 15;
    const int aK = (lane < 16) ? 0 : 16;
    const int bK = (lane < 16) ? 0 : 16;

    if (wave == 0)
        tdm_load_2d(&As[0], A + (size_t)mb * K1P, K1P, 16, K1P);

    // prefetch first B tiles from global while the TDM transfer is in flight
    const __bf16* bp = Bw + (nb + nc) * K1P + bK;
    v16bf b0c = load2x8(bp,            bp + 8);
    v16bf b1c = load2x8(bp + 16*K1P,   bp + 16*K1P + 8);
    v16bf b2c = load2x8(bp + 32*K1P,   bp + 32*K1P + 8);
    v16bf b3c = load2x8(bp + 48*K1P,   bp + 48*K1P + 8);

    if (wave == 0) __builtin_amdgcn_s_wait_tensorcnt(0);
    __syncthreads();

    const __bf16* ar = &As[nc * K1P + aK];
    v16bf a_c = load2x8(ar, ar + 8);
    v8f c0 = {0,0,0,0,0,0,0,0}, c1 = c0, c2 = c0, c3 = c0;

    for (int kk = 32; kk < K1P; kk += 32) {
        v16bf a_n  = load2x8(ar + kk, ar + kk + 8);
        v16bf bn0  = load2x8(bp + kk,           bp + kk + 8);
        v16bf bn1  = load2x8(bp + 16*K1P + kk,  bp + 16*K1P + kk + 8);
        v16bf bn2  = load2x8(bp + 32*K1P + kk,  bp + 32*K1P + kk + 8);
        v16bf bn3  = load2x8(bp + 48*K1P + kk,  bp + 48*K1P + kk + 8);
        c0 = WMMA_BF16(a_c, b0c, c0);
        c1 = WMMA_BF16(a_c, b1c, c1);
        c2 = WMMA_BF16(a_c, b2c, c2);
        c3 = WMMA_BF16(a_c, b3c, c3);
        a_c = a_n; b0c = bn0; b1c = bn1; b2c = bn2; b3c = bn3;
    }
    c0 = WMMA_BF16(a_c, b0c, c0);
    c1 = WMMA_BF16(a_c, b1c, c1);
    c2 = WMMA_BF16(a_c, b2c, c2);
    c3 = WMMA_BF16(a_c, b3c, c3);

    const float beta = *beta_p, thr = *thr_p;
    const int mrow = mb + ((lane >> 4) << 3);
    v8f cc[4] = {c0, c1, c2, c3};
    #pragma unroll
    for (int j = 0; j < 4; ++j) {
        const int col = nb + j * 16 + nc;
        const float bv = bias[col];
        #pragma unroll
        for (int r = 0; r < 8; ++r) {
            const int idx = (mrow + r) * H1 + col;
            float m = beta * mem1[idx] + cc[j][r] + bv;
            float s = (m > thr) ? 1.0f : 0.0f;
            mem1[idx] = m - s * thr;
            s1[idx] = (__bf16)s;
        }
    }
}

// Layer 2: cur = [s1 | spk_r_prev] @ [W2|V]^T ; fused recurrent AHP neuron.
// A staged via TDM in 8 chunks of 16xKC, double-buffered in LDS; TDM(c+1)
// overlaps compute(c); B register double-buffered from global (L2-resident).
__global__ __launch_bounds__(128) void gemm2_neuron2(
    const __bf16* __restrict__ s1,     // [BATCH][H1]
    const __bf16* __restrict__ srp,    // [BATCH][H2] previous spikes
    const __bf16* __restrict__ Bw,     // [H2][K2]
    const float*  __restrict__ bias,
    float*        __restrict__ mem_r,
    float*        __restrict__ ahp,
    __bf16*       __restrict__ src,    // current spikes out
    const float*  __restrict__ beta_p,
    const float*  __restrict__ thr_p,
    const float*  __restrict__ bb_p,
    const float*  __restrict__ alpha_p)
{
    __shared__ __align__(16) __bf16 As[2][16 * KC];
    const int lane = threadIdx.x & 31;
    const int wave = threadIdx.x >> 5;
    const int mb = blockIdx.x * 16;
    const int nb = blockIdx.y * 256 + wave * 64;
    const int nc = lane & 15;
    const int aK = (lane < 16) ? 0 : 16;
    const int bK = (lane < 16) ? 0 : 16;

    const __bf16* a_src0 = s1  + (size_t)mb * H1;
    const __bf16* a_src1 = srp + (size_t)mb * H2;
    const __bf16* bp = Bw + (nb + nc) * K2 + bK;

    if (wave == 0) {
        tdm_load_2d(&As[0][0], a_src0, KC, 16, H1);
        __builtin_amdgcn_s_wait_tensorcnt(0);
    }
    __syncthreads();

    v8f c0 = {0,0,0,0,0,0,0,0}, c1 = c0, c2 = c0, c3 = c0;

    for (int c = 0; c < K2 / KC; ++c) {
        if (wave == 0 && c + 1 < K2 / KC) {
            const int cn = c + 1;
            const __bf16* g = (cn < H1 / KC) ? (a_src0 + cn * KC)
                                             : (a_src1 + (cn - H1 / KC) * KC);
            tdm_load_2d(&As[(c + 1) & 1][0], g, KC, 16, H1);
        }
        const __bf16* ar = &As[c & 1][nc * KC + aK];
        const int kb = c * KC;

        v16bf a_c = load2x8(ar, ar + 8);
        v16bf b0c = load2x8(bp + kb,           bp + kb + 8);
        v16bf b1c = load2x8(bp + 16*K2 + kb,   bp + 16*K2 + kb + 8);
        v16bf b2c = load2x8(bp + 32*K2 + kb,   bp + 32*K2 + kb + 8);
        v16bf b3c = load2x8(bp + 48*K2 + kb,   bp + 48*K2 + kb + 8);

        for (int kk = 32; kk < KC; kk += 32) {
            const int kg = kb + kk;
            v16bf a_n = load2x8(ar + kk, ar + kk + 8);
            v16bf bn0 = load2x8(bp + kg,           bp + kg + 8);
            v16bf bn1 = load2x8(bp + 16*K2 + kg,   bp + 16*K2 + kg + 8);
            v16bf bn2 = load2x8(bp + 32*K2 + kg,   bp + 32*K2 + kg + 8);
            v16bf bn3 = load2x8(bp + 48*K2 + kg,   bp + 48*K2 + kg + 8);
            c0 = WMMA_BF16(a_c, b0c, c0);
            c1 = WMMA_BF16(a_c, b1c, c1);
            c2 = WMMA_BF16(a_c, b2c, c2);
            c3 = WMMA_BF16(a_c, b3c, c3);
            a_c = a_n; b0c = bn0; b1c = bn1; b2c = bn2; b3c = bn3;
        }
        c0 = WMMA_BF16(a_c, b0c, c0);
        c1 = WMMA_BF16(a_c, b1c, c1);
        c2 = WMMA_BF16(a_c, b2c, c2);
        c3 = WMMA_BF16(a_c, b3c, c3);

        if (wave == 0) __builtin_amdgcn_s_wait_tensorcnt(0);  // TDM(c+1) done
        __syncthreads();
    }

    const float beta = *beta_p, thr = *thr_p, bb = *bb_p, al = *alpha_p;
    const int mrow = mb + ((lane >> 4) << 3);
    v8f cc[4] = {c0, c1, c2, c3};
    #pragma unroll
    for (int j = 0; j < 4; ++j) {
        const int col = nb + j * 16 + nc;
        const float bv = bias[col];
        #pragma unroll
        for (int r = 0; r < 8; ++r) {
            const int idx = (mrow + r) * H2 + col;
            const float a_old = ahp[idx];
            float m = beta * mem_r[idx] + (cc[j][r] + bv) - a_old;
            float s = (m > thr) ? 1.0f : 0.0f;
            mem_r[idx] = m - s * thr;
            ahp[idx]   = bb * a_old + al * s;
            src[idx]   = (__bf16)s;
        }
    }
}

// Layer 3: tiny GEMM (K=2048, N=20) + Leaky output neuron. One wave per (b,o).
__global__ __launch_bounds__(256) void layer3(
    const __bf16* __restrict__ sr,
    const float*  __restrict__ W3,
    const float*  __restrict__ b3,
    float*        __restrict__ mem2,
    float*        __restrict__ out_t,
    const float*  __restrict__ beta_p,
    const float*  __restrict__ thr_p)
{
    const int gw = blockIdx.x * 8 + (threadIdx.x >> 5);
    const int lane = threadIdx.x & 31;
    if (gw >= BATCH * NOUT) return;
    const int b = gw / NOUT, o = gw % NOUT;
    const __bf16* s = sr + b * H2;
    const float*  w = W3 + o * H2;
    float acc = 0.0f;
    for (int k = lane; k < H2; k += 32) acc += (float)s[k] * w[k];
    #pragma unroll
    for (int off = 16; off > 0; off >>= 1) acc += __shfl_down(acc, off, 32);
    if (lane == 0) {
        float m = (*beta_p) * mem2[b * NOUT + o] + acc + b3[o];
        float sp = (m > *thr_p) ? 1.0f : 0.0f;
        mem2[b * NOUT + o] = m - sp * (*thr_p);
        out_t[b * NOUT + o] = sp;
    }
}

// ---------------- host launcher ----------------

extern "C" void kernel_launch(void* const* d_in, const int* in_sizes, int n_in,
                              void* d_out, int out_size, void* d_ws, size_t ws_size,
                              hipStream_t stream) {
    const float* data  = (const float*)d_in[0];
    const float* W1    = (const float*)d_in[1];
    const float* b1    = (const float*)d_in[2];
    const float* W2    = (const float*)d_in[3];
    const float* b2    = (const float*)d_in[4];
    const float* V     = (const float*)d_in[5];
    const float* W3    = (const float*)d_in[6];
    const float* b3    = (const float*)d_in[7];
    const float* beta1 = (const float*)d_in[8];
    const float* thr1  = (const float*)d_in[9];
    const float* betar = (const float*)d_in[10];
    const float* thrr  = (const float*)d_in[11];
    const float* backb = (const float*)d_in[12];
    const float* alpha = (const float*)d_in[13];
    const float* beta2 = (const float*)d_in[14];
    const float* thr2  = (const float*)d_in[15];
    float* out = (float*)d_out;

    char* w = (char*)d_ws;
    __bf16* xT   = (__bf16*)w;
    size_t off = (size_t)T_STEPS * BATCH * K1P * 2;
    __bf16* W1b  = (__bf16*)(w + off);  off += (size_t)H1 * K1P * 2;
    __bf16* W2Vb = (__bf16*)(w + off);  off += (size_t)H2 * K2  * 2;
    float* state = (float*)(w + off);
    float* mem1  = state;
    float* mem_r = mem1  + BATCH * H1;
    float* ahp   = mem_r + BATCH * H2;
    float* mem2  = ahp   + BATCH * H2;
    __bf16* sr0  = (__bf16*)(mem2 + BATCH * NOUT);
    __bf16* sr1  = sr0 + BATCH * H2;
    __bf16* s1   = sr1 + BATCH * H2;
    const long state_words = (long)BATCH * H1 + 2L * BATCH * H2 + BATCH * NOUT
                           + (long)BATCH * H2;  // two bf16 spike slots as words

    zero_f32<<<2048, 256, 0, stream>>>(state, state_words);
    cvt_w1 <<<(H1 * K1P + 255) / 256, 256, 0, stream>>>(W1, W1b);
    cvt_w2v<<<(H2 * K2 + 255) / 256, 256, 0, stream>>>(W2, V, W2Vb);
    {
        long n = (long)T_STEPS * BATCH * K1P;
        cvt_x<<<(int)((n + 255) / 256), 256, 0, stream>>>(data, xT);
    }

    dim3 gemm_grid(BATCH / 16, H1 / 256);
    for (int t = 0; t < T_STEPS; ++t) {
        __bf16* sr_cur  = (t & 1) ? sr1 : sr0;
        __bf16* sr_prev = (t & 1) ? sr0 : sr1;   // t=0 reads zeroed sr1
        gemm1_neuron1<<<gemm_grid, 128, 0, stream>>>(
            xT + (size_t)t * BATCH * K1P, W1b, b1, mem1, s1, beta1, thr1);
        gemm2_neuron2<<<gemm_grid, 128, 0, stream>>>(
            s1, sr_prev, W2Vb, b2, mem_r, ahp, sr_cur, betar, thrr, backb, alpha);
        layer3<<<(BATCH * NOUT) / 8, 256, 0, stream>>>(
            sr_cur, W3, b3, mem2, out + (size_t)t * BATCH * NOUT, beta2, thr2);
    }
}
